// MS_BnnConv_75368086110750
// MI455X (gfx1250) — compile-verified, matrix operand order checked by reading, not run on hardware
//
#include <hip/hip_runtime.h>
#include <hip/hip_bf16.h>

typedef _Float16 f16;
typedef __attribute__((ext_vector_type(16))) _Float16 v16h;
typedef __attribute__((ext_vector_type(8)))  float    v8f;

// ---- problem dims ----
#define DT 4
#define DB 8
#define CIN 256
#define DH 64
#define DW 64
#define NN (DT*DB)          // 32
#define C1 192              // conv1 out channels
#define C2 320              // conv2 out channels
#define HWSZ (DH*DW)
#define LSTRIDE 272         // padded per-pixel LDS stride in halves (272*2=544B, 32B aligned, bank-skewed)
#define EPSF 1e-5f

// =====================================================================
// LIF: v = v + (x - v)/2 ; s = (v-1 >= 0) ; v -= s. Spikes -> NHWC f16.
// =====================================================================
__global__ void lif_kernel(const float* __restrict__ x, f16* __restrict__ sp) {
    int gid = blockIdx.x * blockDim.x + threadIdx.x;
    const int total = DB * CIN * HWSZ;
    if (gid >= total) return;
    int b  = gid / (CIN * HWSZ);
    int c  = (gid / HWSZ) % CIN;
    int hw = gid % HWSZ;
    int h = hw >> 6, w = hw & 63;
    float v = 0.f;
#pragma unroll
    for (int t = 0; t < DT; ++t) {
        float xv = x[((t * DB + b) * CIN + c) * HWSZ + hw];
        v += (xv - v) * 0.5f;
        float s = (v - 1.0f >= 0.f) ? 1.f : 0.f;
        v -= s;
        sp[(((t * DB + b) * DH + h) * DW + w) * CIN + c] = (f16)s;
    }
}

// =====================================================================
// Repack conv1 weights: OIHW f32 -> [tap][co][ci] f16
// =====================================================================
__global__ void prep_w1(const float* __restrict__ w1, f16* __restrict__ w1h) {
    int i = blockIdx.x * blockDim.x + threadIdx.x;
    if (i >= 9 * C1 * CIN) return;
    int tap = i / (C1 * CIN);
    int rem = i - tap * (C1 * CIN);
    int co = rem / CIN, ci = rem % CIN;
    w1h[(tap * C1 + co) * CIN + ci] = (f16)w1[(co * CIN + ci) * 9 + tap];
}

// =====================================================================
// Repack conv2 weights: sign(OIHW f32) -> [tap][co][ci64] f16, 48->64 zero pad
// =====================================================================
__global__ void prep_w2(const float* __restrict__ w2, f16* __restrict__ w2h) {
    int i = blockIdx.x * blockDim.x + threadIdx.x;
    if (i >= 9 * C2 * 64) return;
    int tap = i / (C2 * 64);
    int rem = i - tap * (C2 * 64);
    int co = rem / 64, k = rem % 64;
    float s = 0.f;
    if (k < 48) {
        float v = w2[(co * 48 + k) * 9 + tap];
        s = (v > 0.f) ? 1.f : ((v < 0.f) ? -1.f : 0.f);
    }
    w2h[(tap * C2 + co) * 64 + k] = (f16)s;
}

// =====================================================================
// conv1: spikes(NHWC f16) x w1h -> y (NHWC f32) + BN sum/sumsq atomics.
// One workgroup (256 thr = 8 waves) per (n, h): 192co x 64px output tile.
// Wave grid: 4 co-waves (48 co = 3 frags) x 2 px-waves (32 px = 2 frags).
// K = 9 taps x 8 chunks of 32 ch, v_wmma_f32_16x16x32_f16.
// Staging uses GLOBAL_LOAD_ASYNC_TO_LDS_B128 (ASYNCcnt) for the valid
// 64-px spans; halo columns / out-of-range rows zeroed with ds stores.
// Fragment channel map: ch = base + 16*half + e for both A and B (the HW
// multiplies A(e,half) with B(e,half), so any consistent bijection works) ->
// every fragment is one contiguous 32B load per lane.
// =====================================================================
__global__ void __launch_bounds__(256)
conv1_kernel(const f16* __restrict__ sp, const f16* __restrict__ w1h,
             float* __restrict__ y, float* __restrict__ gsum, float* __restrict__ gsq) {
    extern __shared__ __align__(32) char smem_raw[];
    f16* lds = (f16*)smem_raw;
    const int tid = threadIdx.x;
    const int n = blockIdx.x >> 6;
    const int h = blockIdx.x & 63;
    const uint4 z4 = {0u, 0u, 0u, 0u};

    // 1) zero halo columns p=0 and p=65 (3 rows x 2 px x 256ch = 192 x 16B)
    for (int i = tid; i < 3 * 2 * 32; i += 256) {
        int r = i / 64;
        int rem = i - r * 64;
        int p = (rem < 32) ? 0 : 65;
        int c16 = rem & 31;
        *(uint4*)(lds + (r * 66 + p) * LSTRIDE + c16 * 8) = z4;
    }
    // 2) zero out-of-range rows (image-edge padding)
    if (h == 0)
        for (int i = tid; i < 2048; i += 256)
            *(uint4*)(lds + (0 * 66 + 1 + (i >> 5)) * LSTRIDE + (i & 31) * 8) = z4;
    if (h == DH - 1)
        for (int i = tid; i < 2048; i += 256)
            *(uint4*)(lds + (2 * 66 + 1 + (i >> 5)) * LSTRIDE + (i & 31) * 8) = z4;
    // 3) async copy valid rows straight into LDS (no VGPR round-trip)
#pragma unroll
    for (int r = 0; r < 3; ++r) {
        int row = h + r - 1;
        if (row < 0 || row >= DH) continue;
        for (int j = tid; j < 2048; j += 256) {
            int p = j >> 5, c16 = j & 31;
            unsigned long long ga = (unsigned long long)(uintptr_t)
                (sp + ((n * DH + row) * DW + p) * CIN + c16 * 8);
            unsigned la = (unsigned)(((r * 66 + p + 1) * LSTRIDE + c16 * 8) * 2);
            asm volatile("global_load_async_to_lds_b128 %0, %1, off"
                         :: "v"(la), "v"(ga) : "memory");
        }
    }
    asm volatile("s_wait_asynccnt 0x0" ::: "memory");
    __syncthreads();

    const int lane = tid & 31, half = lane >> 4, l16 = lane & 15;
    const int wave = tid >> 5;
    const int coB  = (wave & 3) * 48;
    const int pixB = (wave >> 2) * 32;

    const v8f z8 = {0.f,0.f,0.f,0.f,0.f,0.f,0.f,0.f};
    v8f acc[3][2];
#pragma unroll
    for (int cf = 0; cf < 3; ++cf)
#pragma unroll
        for (int pf = 0; pf < 2; ++pf) acc[cf][pf] = z8;

    for (int tap = 0; tap < 9; ++tap) {
        const int kh = tap / 3, kw = tap % 3;
        for (int ck = 0; ck < 8; ++ck) {
            const int base = ck * 32;
            v16h a[3], bb[2];
#pragma unroll
            for (int cf = 0; cf < 3; ++cf) {
                int co = coB + cf * 16 + l16;
                a[cf] = *(const v16h*)(w1h + (tap * C1 + co) * CIN + base + 16 * half);
            }
#pragma unroll
            for (int pf = 0; pf < 2; ++pf) {
                int pix = pixB + pf * 16 + l16;
                bb[pf] = *(const v16h*)(lds + (kh * 66 + pix + kw) * LSTRIDE + base + 16 * half);
            }
#pragma unroll
            for (int cf = 0; cf < 3; ++cf)
#pragma unroll
                for (int pf = 0; pf < 2; ++pf)
                    acc[cf][pf] = __builtin_amdgcn_wmma_f32_16x16x32_f16(
                        false, a[cf], false, bb[pf], (short)0, acc[cf][pf], false, false);
        }
    }

    // ---- store y (NHWC f32): lane l16 = pixel N, vgpr v + 8*half = co row M ----
#pragma unroll
    for (int cf = 0; cf < 3; ++cf)
#pragma unroll
        for (int pf = 0; pf < 2; ++pf)
#pragma unroll
            for (int v = 0; v < 8; ++v) {
                int co  = coB + cf * 16 + v + 8 * half;
                int pix = pixB + pf * 16 + l16;
                y[((n * DH + h) * DW + pix) * C1 + co] = acc[cf][pf][v];
            }

    // ---- BN partial sums: shfl-reduce 16 px lanes, LDS atomics, 1 global atomic/ch ----
    __syncthreads();
    float* lsum = (float*)smem_raw;       // 192 floats
    float* lsq  = lsum + 192;             // 192 floats (contiguous)
    for (int i = tid; i < 384; i += 256) lsum[i] = 0.f;
    __syncthreads();
#pragma unroll
    for (int cf = 0; cf < 3; ++cf)
#pragma unroll
        for (int v = 0; v < 8; ++v) {
            float a0 = acc[cf][0][v], a1 = acc[cf][1][v];
            float s = a0 + a1;
            float q = a0 * a0 + a1 * a1;
#pragma unroll
            for (int m = 1; m < 16; m <<= 1) {
                s += __shfl_xor(s, m, 32);
                q += __shfl_xor(q, m, 32);
            }
            if (l16 == 0) {
                int co = coB + cf * 16 + v + 8 * half;
                atomicAdd(&lsum[co], s);
                atomicAdd(&lsq[co], q);
            }
        }
    __syncthreads();
    if (tid < C1) {
        atomicAdd(&gsum[tid], lsum[tid]);
        atomicAdd(&gsq[tid], lsq[tid]);
    }
}

// =====================================================================
// conv2: BN(y) staged to LDS as f16 (groups padded 48->64 ch), grouped
// binary-weight conv via WMMA; writes x1 (ch 0..191) and x2 (ch 192..511)
// of the NCHW output. Wave w: group g = w&3 (80 co = 5 frags), px half.
// =====================================================================
__global__ void __launch_bounds__(256)
conv2_kernel(const float* __restrict__ y, const f16* __restrict__ w2h,
             const float* __restrict__ gsum, const float* __restrict__ gsq,
             const float* __restrict__ gamma, const float* __restrict__ beta,
             float* __restrict__ out) {
    extern __shared__ __align__(32) char smem_raw[];
    f16* lds = (f16*)smem_raw;
    __shared__ float bnS[C1], bnB[C1];
    const int tid = threadIdx.x;
    const int n = blockIdx.x >> 6;
    const int h = blockIdx.x & 63;

    if (tid < C1) {
        const float cnt = (float)(NN * HWSZ);
        float mean = gsum[tid] / cnt;
        float var  = gsq[tid] / cnt - mean * mean;
        float sc   = gamma[tid] * rsqrtf(var + EPSF);
        bnS[tid] = sc;
        bnB[tid] = beta[tid] - mean * sc;
    }
    __syncthreads();

    // zero LDS (covers halo + the 48->64 channel padding)
    for (int i = tid; i < 3 * 66 * LSTRIDE / 2; i += 256) ((unsigned*)lds)[i] = 0u;
    __syncthreads();

    // stage BN(y) as f16, channel c -> lds slot (c/48)*64 + c%48
    for (int idx = tid; idx < 3 * 66 * C1; idx += 256) {
        int r   = idx / (66 * C1);
        int rem = idx - r * (66 * C1);
        int p   = rem / C1;
        int c   = rem - p * C1;
        int row = h + r - 1;
        if (row >= 0 && row < DH && p >= 1 && p <= DW) {
            float val = y[((n * DH + row) * DW + (p - 1)) * C1 + c] * bnS[c] + bnB[c];
            lds[(r * 66 + p) * LSTRIDE + (c / 48) * 64 + (c % 48)] = (f16)val;
        }
    }
    __syncthreads();

    // x1: BN(y) in full f32, NCHW channels 0..191 (coalesced along w)
    for (int idx = tid; idx < C1 * DW; idx += 256) {
        int c = idx >> 6, w = idx & 63;
        float val = y[((n * DH + h) * DW + w) * C1 + c] * bnS[c] + bnB[c];
        out[((n * 512 + c) * DH + h) * DW + w] = val;
    }

    const int lane = tid & 31, half = lane >> 4, l16 = lane & 15;
    const int wave = tid >> 5;
    const int g    = wave & 3;          // group: 80 out-ch, 64 (padded) in-ch
    const int pixB = (wave >> 2) * 32;

    const v8f z8 = {0.f,0.f,0.f,0.f,0.f,0.f,0.f,0.f};
    v8f acc[5][2];
#pragma unroll
    for (int cf = 0; cf < 5; ++cf)
#pragma unroll
        for (int pf = 0; pf < 2; ++pf) acc[cf][pf] = z8;

    for (int tap = 0; tap < 9; ++tap) {
        const int kh = tap / 3, kw = tap % 3;
#pragma unroll
        for (int ck = 0; ck < 2; ++ck) {
            v16h a[5], bb[2];
#pragma unroll
            for (int cf = 0; cf < 5; ++cf) {
                int co = g * 80 + cf * 16 + l16;
                a[cf] = *(const v16h*)(w2h + (tap * C2 + co) * 64 + ck * 32 + 16 * half);
            }
#pragma unroll
            for (int pf = 0; pf < 2; ++pf) {
                int pix = pixB + pf * 16 + l16;
                bb[pf] = *(const v16h*)(lds + (kh * 66 + pix + kw) * LSTRIDE + g * 64 + ck * 32 + 16 * half);
            }
#pragma unroll
            for (int cf = 0; cf < 5; ++cf)
#pragma unroll
                for (int pf = 0; pf < 2; ++pf)
                    acc[cf][pf] = __builtin_amdgcn_wmma_f32_16x16x32_f16(
                        false, a[cf], false, bb[pf], (short)0, acc[cf][pf], false, false);
        }
    }

    // x2: NCHW channels 192..511
#pragma unroll
    for (int cf = 0; cf < 5; ++cf)
#pragma unroll
        for (int pf = 0; pf < 2; ++pf)
#pragma unroll
            for (int v = 0; v < 8; ++v) {
                int co  = g * 80 + cf * 16 + v + 8 * half;
                int pix = pixB + pf * 16 + l16;
                out[((n * 512 + 192 + co) * DH + h) * DW + pix] = acc[cf][pf][v];
            }
}

// =====================================================================
// Host-side launch. Workspace layout (bytes):
//   [0)            spikes f16  NHWC  : 32*64*64*256*2  =  67,108,864
//   [67108864)     y f32       NHWC  : 32*64*64*192*4  = 100,663,296
//   [167772160)    w1h f16 [9][192][256]               =     884,736
//   [168656896)    w2h f16 [9][320][64]                =     368,640
//   [169025536)    gsum f32[192], gsq f32[192]         =       1,536
//   total ~161.2 MB (must be <= ws_size)
// =====================================================================
extern "C" void kernel_launch(void* const* d_in, const int* in_sizes, int n_in,
                              void* d_out, int out_size, void* d_ws, size_t ws_size,
                              hipStream_t stream) {
    const float* x     = (const float*)d_in[0];
    const float* w1    = (const float*)d_in[1];
    const float* gamma = (const float*)d_in[2];
    const float* beta  = (const float*)d_in[3];
    const float* w2    = (const float*)d_in[4];
    float* out = (float*)d_out;

    char* ws   = (char*)d_ws;
    f16*   sp   = (f16*)(ws);
    float* y    = (float*)(ws + 67108864);
    f16*   w1h  = (f16*)(ws + 167772160);
    f16*   w2h  = (f16*)(ws + 168656896);
    float* gsum = (float*)(ws + 169025536);
    float* gsq  = gsum + C1;

    hipMemsetAsync(gsum, 0, 2 * C1 * sizeof(float), stream);

    {
        int total = DB * CIN * HWSZ;
        lif_kernel<<<(total + 255) / 256, 256, 0, stream>>>(x, sp);
    }
    prep_w1<<<(9 * C1 * CIN + 255) / 256, 256, 0, stream>>>(w1, w1h);
    prep_w2<<<(9 * C2 * 64 + 255) / 256, 256, 0, stream>>>(w2, w2h);

    const size_t ldsBytes = (size_t)(3 * 66 * LSTRIDE) * sizeof(f16); // 107,712 B
    conv1_kernel<<<NN * DH, 256, ldsBytes, stream>>>(sp, w1h, y, gsum, gsq);
    conv2_kernel<<<NN * DH, 256, ldsBytes, stream>>>(y, w2h, gsum, gsq, gamma, beta, out);
}
